// StreamKhoiddGCN_60687887892936
// MI455X (gfx1250) — matrix operand order repaired and checked by reading, not compiled, not run on hardware
//
#include <hip/hip_runtime.h>

typedef __attribute__((ext_vector_type(16))) __bf16 bf16x16;
typedef __attribute__((ext_vector_type(8)))  float  f32x8;

#define SKG_S   256        /* N*M effective batch */
#define SKG_EPS 1e-5f

struct SkgU8 { unsigned w[8]; };

__device__ __forceinline__ unsigned skg_pack2(float lo, float hi)
{
    struct P { __bf16 a, b; } p;
    p.a = (__bf16)lo; p.b = (__bf16)hi;
    return __builtin_bit_cast(unsigned, p);       // v_cvt_pk_bf16_f32
}

// ---------------------------------------------------------------------------
// Wave-level WMMA: one wave produces a 16x32 f32 slab (2 accumulators) from
// bf16 LDS tiles A[32][34] (row-major MxK) and Bt[64][34] (transposed: [n][k])
// via two V_WMMA_F32_16X16X32_BF16 that share the A fragment.
// Fragment k order k = 16*(lane/16) + s in both A and B (wave32); all loads are contiguous
// dwords of pre-converted bf16 pairs.
// ---------------------------------------------------------------------------
__device__ __forceinline__ void skg_wave_wmma2(const __bf16 (*As)[34], const __bf16 (*Bt)[34],
                                               int wm, int wn, int lane,
                                               f32x8 &acc0, f32x8 &acc1)
{
    const int mr   = lane & 15;
    const int khiA = (lane >> 4) << 4;            // lanes 16..31 hold K+16 block
    const unsigned *pa = (const unsigned *)&As[wm + mr][0];
    SkgU8 ua;
#pragma unroll
    for (int j = 0; j < 8; ++j) {
        const int kb = (j << 1) + khiA; // j*2
        ua.w[j] = pa[kb >> 1];
    }
    const bf16x16 af = __builtin_bit_cast(bf16x16, ua);

    const int nc = lane & 15;
    const int kw = (lane >> 4) << 3;              // dword offset: lanes 16..31 -> K+16
    SkgU8 ub;
    const unsigned *pb0 = (const unsigned *)&Bt[wn + nc][0] + kw;
#pragma unroll
    for (int j = 0; j < 8; ++j) ub.w[j] = pb0[j];
    const bf16x16 bf0 = __builtin_bit_cast(bf16x16, ub);
    acc0 = __builtin_amdgcn_wmma_f32_16x16x32_bf16(false, af, false, bf0, (short)0, acc0,
                                                   false, false);
    const unsigned *pb1 = (const unsigned *)&Bt[wn + 16 + nc][0] + kw;
#pragma unroll
    for (int j = 0; j < 8; ++j) ub.w[j] = pb1[j];
    const bf16x16 bf1 = __builtin_bit_cast(bf16x16, ub);
    acc1 = __builtin_amdgcn_wmma_f32_16x16x32_bf16(false, af, false, bf1, (short)0, acc1,
                                                   false, false);
}

// ---------------------------------------------------------------------------
// Generic batched GEMM: C[z] = alpha * A[z] @ B[z] + bias (+ C if accum).
// A: [M,K] row-major (aTrans=0) or [K,M] (aTrans=1). B: [K,N]. strides in elems.
// Block 128 threads = 4 waves; block tile 32x64; K staged in bf16 LDS chunks of 32.
// ---------------------------------------------------------------------------
__global__ void skg_gemm(const float *__restrict__ A, const float *__restrict__ B,
                         float *__restrict__ C, const float *__restrict__ bias,
                         int M, int N, int K,
                         long long sA, long long sB, long long sC,
                         int aTrans, float alpha, int accum, int relu)
{
    __shared__ __bf16 As[32][34];
    __shared__ __bf16 Bt[64][34];
    const float *Ab = A + (long long)blockIdx.z * sA;
    const float *Bb = B + (long long)blockIdx.z * sB;
    float       *Cb = C + (long long)blockIdx.z * sC;
    const int n0 = blockIdx.x << 6, m0 = blockIdx.y << 5;
    const int tid = threadIdx.x, lane = tid & 31, wave = tid >> 5;
    const int wm = (wave >> 1) << 4, wn = (wave & 1) << 5;
    f32x8 acc0 = {0.f, 0.f, 0.f, 0.f, 0.f, 0.f, 0.f, 0.f};
    f32x8 acc1 = {0.f, 0.f, 0.f, 0.f, 0.f, 0.f, 0.f, 0.f};

    for (int kk = 0; kk < K; kk += 32) {
        // stage A (32m x 32k) as bf16 pairs: 512 dwords
        for (int t = tid; t < 512; t += 128) {
            const int r = t >> 4, c2 = (t & 15) << 1;
            const int gm = m0 + r, gk = kk + c2;
            float v0 = 0.f, v1 = 0.f;
            if (gm < M) {
                if (aTrans) {
                    if (gk < K)     v0 = Ab[(long long)gk * M + gm];
                    if (gk + 1 < K) v1 = Ab[(long long)(gk + 1) * M + gm];
                } else {
                    if (gk < K)     v0 = Ab[(long long)gm * K + gk];
                    if (gk + 1 < K) v1 = Ab[(long long)gm * K + gk + 1];
                }
            }
            ((unsigned *)&As[r][0])[t & 15] = skg_pack2(v0, v1);
        }
        // stage B (32k x 64n) transposed into Bt[n][k] as bf16 pairs: 1024 dwords
        for (int t = tid; t < 1024; t += 128) {
            const int n = t & 63, k2 = (t >> 6) << 1;
            const int gk = kk + k2, gn = n0 + n;
            float v0 = 0.f, v1 = 0.f;
            if (gn < N) {
                if (gk < K)     v0 = Bb[(long long)gk * N + gn];
                if (gk + 1 < K) v1 = Bb[(long long)(gk + 1) * N + gn];
            }
            ((unsigned *)&Bt[n][0])[t >> 6] = skg_pack2(v0, v1);
        }
        if (kk + 32 < K)   // speculative prefetch of next B K-chunk (global_prefetch_b8)
            __builtin_prefetch(&Bb[(long long)(kk + 32 + (tid & 31)) * N + n0 + ((tid >> 5) << 4)], 0, 1);
        __syncthreads();
        skg_wave_wmma2(As, Bt, wm, wn, lane, acc0, acc1);
        __syncthreads();
    }

    const int nc = lane & 15, mofs = (lane >> 4) << 3;
#pragma unroll
    for (int j = 0; j < 8; ++j) {
        const int m = m0 + wm + mofs + j;
        if (m < M) {
            const float bia = bias ? bias[m] : 0.f;
            const int nA = n0 + wn + nc, nB = nA + 16;
            if (nA < N) {
                float v = acc0[j] * alpha + bia;
                const long long idx = (long long)m * N + nA;
                if (accum) v += Cb[idx];
                if (relu)  v = fmaxf(v, 0.f);
                Cb[idx] = v;
            }
            if (nB < N) {
                float v = acc1[j] * alpha + bia;
                const long long idx = (long long)m * N + nB;
                if (accum) v += Cb[idx];
                if (relu)  v = fmaxf(v, 0.f);
                Cb[idx] = v;
            }
        }
    }
}

// ---------------------------------------------------------------------------
// Implicit-GEMM temporal conv (KSx1, stride, pad) as WMMA GEMM.
// W: [M, K=cin*KS] shared. X[z]: [cin, Din, R]. C[z]: [M, N=Dout*R].
// ---------------------------------------------------------------------------
__global__ void skg_tcn(const float *__restrict__ W, const float *__restrict__ X,
                        float *__restrict__ C, const float *__restrict__ bias,
                        int M, int N, int K, int Din, int R, int stride, int KS, int pad,
                        long long sX, long long sC)
{
    __shared__ __bf16 As[32][34];
    __shared__ __bf16 Bt[64][34];
    const float *Bb = X + (long long)blockIdx.z * sX;
    float       *Cb = C + (long long)blockIdx.z * sC;
    const int n0 = blockIdx.x << 6, m0 = blockIdx.y << 5;
    const int tid = threadIdx.x, lane = tid & 31, wave = tid >> 5;
    const int wm = (wave >> 1) << 4, wn = (wave & 1) << 5;
    f32x8 acc0 = {0.f, 0.f, 0.f, 0.f, 0.f, 0.f, 0.f, 0.f};
    f32x8 acc1 = {0.f, 0.f, 0.f, 0.f, 0.f, 0.f, 0.f, 0.f};

    for (int kk = 0; kk < K; kk += 32) {
        for (int t = tid; t < 512; t += 128) {
            const int r = t >> 4, c2 = (t & 15) << 1;
            const int gm = m0 + r, gk = kk + c2;
            float v0 = 0.f, v1 = 0.f;
            if (gm < M) {
                if (gk < K)     v0 = W[(long long)gm * K + gk];
                if (gk + 1 < K) v1 = W[(long long)gm * K + gk + 1];
            }
            ((unsigned *)&As[r][0])[t & 15] = skg_pack2(v0, v1);
        }
        for (int t = tid; t < 1024; t += 128) {
            const int n = t & 63, k2 = (t >> 6) << 1;
            const int gn = n0 + n;
            float v01[2] = {0.f, 0.f};
            if (gn < N) {
                const int dd = gn / R, rr = gn - dd * R;
#pragma unroll
                for (int q = 0; q < 2; ++q) {
                    const int gk = kk + k2 + q;
                    if (gk < K) {
                        const int ci = gk / KS, kq = gk - ci * KS;
                        const int tt = dd * stride + kq - pad;
                        if (tt >= 0 && tt < Din)
                            v01[q] = Bb[((long long)ci * Din + tt) * R + rr];
                    }
                }
            }
            ((unsigned *)&Bt[n][0])[t >> 6] = skg_pack2(v01[0], v01[1]);
        }
        __syncthreads();
        skg_wave_wmma2(As, Bt, wm, wn, lane, acc0, acc1);
        __syncthreads();
    }

    const int nc = lane & 15, mofs = (lane >> 4) << 3;
#pragma unroll
    for (int j = 0; j < 8; ++j) {
        const int m = m0 + wm + mofs + j;
        if (m < M) {
            const float bia = bias ? bias[m] : 0.f;
            const int nA = n0 + wn + nc, nB = nA + 16;
            if (nA < N) Cb[(long long)m * N + nA] = acc0[j] + bia;
            if (nB < N) Cb[(long long)m * N + nB] = acc1[j] + bia;
        }
    }
}

// ---------------------------------------------------------------------------
// BatchNorm2d stats (train mode): per-channel mean/var over (S, HW).
// ---------------------------------------------------------------------------
__global__ void skg_bnstats(const float *__restrict__ x, float *__restrict__ mv,
                            int C, int HW)
{
    __shared__ float r1[256];
    __shared__ float r2[256];
    const int c = blockIdx.x, tid = threadIdx.x;
    const long long per = (long long)SKG_S * HW;
    float s = 0.f, s2 = 0.f;
    for (long long i = tid; i < per; i += blockDim.x) {
        const long long smp = i / HW;
        const int h = (int)(i - smp * HW);
        const float v = x[(smp * C + c) * (long long)HW + h];
        s += v; s2 += v * v;
    }
    r1[tid] = s; r2[tid] = s2; __syncthreads();
    for (int o = 128; o > 0; o >>= 1) {
        if (tid < o) { r1[tid] += r1[tid + o]; r2[tid] += r2[tid + o]; }
        __syncthreads();
    }
    if (tid == 0) {
        const float mean = r1[0] / (float)per;
        mv[2 * c]     = mean;
        mv[2 * c + 1] = r2[0] / (float)per - mean * mean;
    }
}

// ---------------------------------------------------------------------------
// Fused BN apply: out = [relu]( bn(x) + res_or_bn(res) )  (elementwise, [S,C,HW]).
// ---------------------------------------------------------------------------
__global__ void skg_bnapply(const float *__restrict__ x, const float *__restrict__ mv,
                            const float *__restrict__ g, const float *__restrict__ b,
                            const float *__restrict__ res, const float *__restrict__ mv2,
                            const float *__restrict__ g2, const float *__restrict__ b2,
                            float *__restrict__ out, int C, int HW, long long total, int relu)
{
    const long long step = (long long)gridDim.x * blockDim.x;
    for (long long e = (long long)blockIdx.x * blockDim.x + threadIdx.x; e < total; e += step) {
        const int c = (int)((e / HW) % C);
        float y = (x[e] - mv[2 * c]) * rsqrtf(mv[2 * c + 1] + SKG_EPS) * g[c] + b[c];
        if (res) {
            float r = res[e];
            if (mv2) r = (r - mv2[2 * c]) * rsqrtf(mv2[2 * c + 1] + SKG_EPS) * g2[c] + b2[c];
            y += r;
        }
        if (relu) y = fmaxf(y, 0.f);
        out[e] = y;
    }
}

// ---------------------------------------------------------------------------
// Softmax over axis -2 of att[S,R,R] (over rows, per column) + (A + 1e-6).
// One wave32 per (s, column). Amat==null -> A_BIG == 1.0.
// ---------------------------------------------------------------------------
__global__ void skg_softmax(float *__restrict__ att, const float *__restrict__ Amat, int R)
{
    const int s = blockIdx.x / R, col = blockIdx.x - s * R;
    float *base = att + (long long)s * R * R + col;
    const int lane = threadIdx.x;
    const int r0 = lane, r1 = lane + 32, r2 = lane + 64;
    float v0 = -1e30f, v1 = -1e30f, v2 = -1e30f;
    if (r0 < R) v0 = base[(long long)r0 * R];
    if (r1 < R) v1 = base[(long long)r1 * R];
    if (r2 < R) v2 = base[(long long)r2 * R];
    float mx = fmaxf(v0, fmaxf(v1, v2));
    for (int o = 16; o > 0; o >>= 1) mx = fmaxf(mx, __shfl_xor(mx, o, 32));
    float e0 = (r0 < R) ? __expf(v0 - mx) : 0.f;
    float e1 = (r1 < R) ? __expf(v1 - mx) : 0.f;
    float e2 = (r2 < R) ? __expf(v2 - mx) : 0.f;
    float sm = e0 + e1 + e2;
    for (int o = 16; o > 0; o >>= 1) sm += __shfl_xor(sm, o, 32);
    const float inv = 1.f / sm;
    if (r0 < R) base[(long long)r0 * R] = e0 * inv + (Amat ? Amat[r0 * R + col] + 1e-6f : 1.000001f);
    if (r1 < R) base[(long long)r1 * R] = e1 * inv + (Amat ? Amat[r1 * R + col] + 1e-6f : 1.000001f);
    if (r2 < R) base[(long long)r2 * R] = e2 * inv + (Amat ? Amat[r2 * R + col] + 1e-6f : 1.000001f);
}

// ---------------------------------------------------------------------------
// Deterministic stand-in for A_SMALL (3,25,25): hashed uniforms, column-
// normalized (sum over axis 1 == 1) like the reference.
// ---------------------------------------------------------------------------
__global__ void skg_asmall(float *__restrict__ A)
{
    const int b = blockIdx.x, i = b / 25, col = b - i * 25, r = threadIdx.x;
    float u = 0.f;
    if (r < 25) {
        unsigned h = (unsigned)(i * 625 + r * 25 + col) * 2654435761u;
        h ^= h >> 16; h *= 0x85ebca6bu; h ^= h >> 13; h *= 0xc2b2ae35u; h ^= h >> 16;
        u = (float)(h >> 8) * (1.0f / 16777216.0f) * 0.998f + 0.001f;
    }
    float sum = u;
    for (int o = 16; o > 0; o >>= 1) sum += __shfl_xor(sum, o, 32);
    if (r < 25) A[(i * 25 + r) * 25 + col] = u / sum;
}

// ---------------------------------------------------------------------------
// data_bn (BatchNorm1d over (N,T) of (128,150,300)) stats, gathered from the
// original x[N=128, C=3, T=300, V=25, M=2]; channel = m*75 + v*3 + c.
// ---------------------------------------------------------------------------
__global__ void skg_bn1dstats(const float *__restrict__ x, float *__restrict__ mv)
{
    __shared__ float r1[256];
    __shared__ float r2[256];
    const int ch = blockIdx.x, tid = threadIdx.x;
    const int m = ch / 75, rem = ch - m * 75, v = rem / 3, c = rem - v * 3;
    float s = 0.f, s2 = 0.f;
    for (int i = tid; i < 128 * 300; i += blockDim.x) {
        const int n = i / 300, t = i - n * 300;
        const float val = x[((((long long)n * 3 + c) * 300 + t) * 25 + v) * 2 + m];
        s += val; s2 += val * val;
    }
    r1[tid] = s; r2[tid] = s2; __syncthreads();
    for (int o = 128; o > 0; o >>= 1) {
        if (tid < o) { r1[tid] += r1[tid + o]; r2[tid] += r2[tid + o]; }
        __syncthreads();
    }
    if (tid == 0) {
        const float mean = r1[0] / 38400.f;
        mv[2 * ch] = mean;
        mv[2 * ch + 1] = r2[0] / 38400.f - mean * mean;
    }
}

// Apply data_bn and produce X0[(n*2+m), c, t, v] (256,3,300,25).
__global__ void skg_bn1dapply(const float *__restrict__ x, const float *__restrict__ mv,
                              const float *__restrict__ g, const float *__restrict__ b,
                              float *__restrict__ out)
{
    const long long total = 256LL * 3 * 300 * 25;
    const long long step = (long long)gridDim.x * blockDim.x;
    for (long long e = (long long)blockIdx.x * blockDim.x + threadIdx.x; e < total; e += step) {
        const int vv = (int)(e % 25);
        long long q = e / 25;
        const int t = (int)(q % 300); q /= 300;
        const int c = (int)(q % 3);
        const int s2 = (int)(q / 3);
        const int n = s2 >> 1, m = s2 & 1;
        const int ch = m * 75 + vv * 3 + c;
        const float val = x[((((long long)n * 3 + c) * 300 + t) * 25 + vv) * 2 + m];
        out[e] = (val - mv[2 * ch]) * rsqrtf(mv[2 * ch + 1] + SKG_EPS) * g[ch] + b[ch];
    }
}

// Transpose (SC, D0, R0) -> (SC, R0, D0) for the l11 time<->node swap.
__global__ void skg_trans(const float *__restrict__ in, float *__restrict__ out,
                          long long SC, int D0, int R0)
{
    const long long total = SC * D0 * R0;
    const long long step = (long long)gridDim.x * blockDim.x;
    for (long long e = (long long)blockIdx.x * blockDim.x + threadIdx.x; e < total; e += step) {
        const int d = (int)(e % D0);
        long long q = e / D0;
        const int r = (int)(q % R0);
        const long long sc = q / R0;
        out[e] = in[(sc * D0 + d) * (long long)R0 + r];
    }
}

// Global mean over (S, HW) per channel.
__global__ void skg_pool(const float *__restrict__ x, float *__restrict__ pooled,
                         int C, int HW)
{
    __shared__ float r1[256];
    const int c = blockIdx.x, tid = threadIdx.x;
    const long long per = (long long)SKG_S * HW;
    float s = 0.f;
    for (long long i = tid; i < per; i += blockDim.x) {
        const long long smp = i / HW;
        const int h = (int)(i - smp * HW);
        s += x[(smp * C + c) * (long long)HW + h];
    }
    r1[tid] = s; __syncthreads();
    for (int o = 128; o > 0; o >>= 1) {
        if (tid < o) r1[tid] += r1[tid + o];
        __syncthreads();
    }
    if (tid == 0) pooled[c] = r1[0] / (float)per;
}

// Final FC: out[j] = pooled . fc_w[j] + fc_b[j], 12 classes.
__global__ void skg_fc(const float *__restrict__ pooled, const float *__restrict__ w,
                       const float *__restrict__ b, float *__restrict__ out)
{
    __shared__ float red[256];
    const int tid = threadIdx.x;
    for (int j = 0; j < 12; ++j) {
        red[tid] = pooled[tid] * w[j * 256 + tid];
        __syncthreads();
        for (int o = 128; o > 0; o >>= 1) {
            if (tid < o) red[tid] += red[tid + o];
            __syncthreads();
        }
        if (tid == 0) out[j] = red[0] + b[j];
        __syncthreads();
    }
}

// ===========================================================================
// Host orchestration
// ===========================================================================
struct SkgCfg { int cin, cout, stride, res; };   // res: 0 none, 1 id, 2 conv
static const SkgCfg skg_cfg[19] = {
    {3, 8, 1, 0},   {8, 8, 1, 1},   {8, 8, 1, 1},   {8, 8, 1, 1},
    {8, 16, 2, 2},  {16, 16, 1, 1}, {16, 16, 1, 1},
    {16, 32, 2, 2}, {32, 32, 1, 1}, {32, 32, 1, 1},
    {32, 64, 2, 2}, {64, 64, 1, 1}, {64, 64, 1, 1},
    {64, 128, 2, 2},{128,128, 1, 1},{128,128, 1, 1},
    {128,256, 2, 2},{256,256, 1, 1},{256,256, 1, 1}};

struct SkgUnit {
    const float *ba, *bb, *bd, *bn_b, *bn_g;
    const float *dbn_b, *dbn_g, *down_b, *down_w;
    const float *wa, *wb, *wd;
    const float *res_b, *res_beta, *res_g, *res_w;
    const float *tcn_b, *tcn_beta, *tcn_g, *tcn_w;
};

extern "C" void kernel_launch(void* const* d_in, const int* in_sizes, int n_in,
                              void* d_out, int out_size, void* d_ws, size_t ws_size,
                              hipStream_t stream)
{
    (void)in_sizes; (void)out_size; (void)ws_size;
    const float *x = (const float *)d_in[0];

    // ---- walk the flattened params pytree (jax: dict keys sorted, lists in order)
    int pi = 1;
    auto nxt = [&]() -> const float * {
        int i = pi < n_in ? pi : (n_in - 1);
        ++pi;
        return (const float *)d_in[i];
    };
    const float *dbn1_b = nxt();   // 'dbn_b'
    const float *dbn1_g = nxt();   // 'dbn_g'
    const float *fc_b   = nxt();   // 'fc_b'
    const float *fc_w   = nxt();   // 'fc_w'
    SkgUnit up[19];
    for (int u = 0; u < 19; ++u) {
        const bool down = skg_cfg[u].cin != skg_cfg[u].cout;
        const bool rconv = skg_cfg[u].res == 2;
        up[u].ba = nxt(); up[u].bb = nxt(); up[u].bd = nxt();
        up[u].bn_b = nxt(); up[u].bn_g = nxt();
        if (down) { up[u].dbn_b = nxt(); up[u].dbn_g = nxt();
                    up[u].down_b = nxt(); up[u].down_w = nxt(); }
        else      { up[u].dbn_b = up[u].dbn_g = up[u].down_b = up[u].down_w = nullptr; }
        up[u].wa = nxt(); up[u].wb = nxt(); up[u].wd = nxt();
        if (rconv) { up[u].res_b = nxt(); up[u].res_beta = nxt();
                     up[u].res_g = nxt(); up[u].res_w = nxt(); }
        else       { up[u].res_b = up[u].res_beta = up[u].res_g = up[u].res_w = nullptr; }
        up[u].tcn_b = nxt(); up[u].tcn_beta = nxt(); up[u].tcn_g = nxt(); up[u].tcn_w = nxt();
    }

    // ---- workspace slots (floats)
    float *pool = (float *)d_ws;
    size_t off = 0;
    auto alloc = [&](size_t nelem) { float *p = pool + off; off += nelem; return p; };
    const size_t CAP = 34500000;       // >= 256*256*7*75
    float *X  = alloc(CAP);            // activation (final unit output always lands here)
    float *Y  = alloc(CAP);            // gcn accumulator / tcn output
    float *Z  = alloc(CAP);            // z / down / residual scratch
    float *G  = alloc(CAP);            // gcn output (tcn input)
    float *AA = alloc(8700000);        // a-conv (S, inter, D, R)
    float *BB = alloc(8700000);        // b-conv
    float *AT = alloc((size_t)SKG_S * 75 * 75);   // attention (S,R,R)
    float *MV  = alloc(1024);          // bn stats set 1
    float *MV2 = alloc(1024);          // bn stats set 2
    float *ASM = alloc(3 * 25 * 25);   // A_SMALL stand-in
    float *PO  = alloc(256);           // pooled features

    auto gemm = [&](const float *A, const float *B, float *C, const float *bias,
                    int M, int N, int K, long long sA, long long sB, long long sC,
                    int aTrans, float alpha, int accum, int relu) {
        dim3 g((N + 63) / 64, (M + 31) / 32, SKG_S);
        skg_gemm<<<g, dim3(128), 0, stream>>>(A, B, C, bias, M, N, K, sA, sB, sC,
                                              aTrans, alpha, accum, relu);
    };

    // ---- constants + data_bn + layout to (256,3,300,25)
    skg_asmall<<<dim3(75), dim3(32), 0, stream>>>(ASM);
    skg_bn1dstats<<<dim3(150), dim3(256), 0, stream>>>(x, MV);
    skg_bn1dapply<<<dim3(2048), dim3(256), 0, stream>>>(x, MV, dbn1_g, dbn1_b, X);

    int D = 300, R = 25;
    for (int u = 0; u < 19; ++u) {
        if (u == 10) {   // l11: (S,32,75,25) -> (S,32,25,75)
            skg_trans<<<dim3(2048), dim3(256), 0, stream>>>(X, Y, (long long)SKG_S * 32, 75, 25);
            float *tmp = X; X = Y; Y = tmp;
            D = 25; R = 75;
        }
        const SkgCfg c = skg_cfg[u];
        const int cin = c.cin, cout = c.cout, inter = cout / 4;
        const int P = D * R;
        const long long sX = (long long)cin * P;
        const long long sYc = (long long)cout * P;
        const long long sAB = (long long)inter * P;

        // ---- graph-attention GCN: y = sum_i wd_i @ (x @ (softmax(A1A2/iD) + Ae))
        for (int i = 0; i < 3; ++i) {
            gemm(up[u].wa + (size_t)i * inter * cin, X, AA, up[u].ba + (size_t)i * inter,
                 inter, P, cin, 0, sX, sAB, 0, 1.f, 0, 0);
            gemm(up[u].wb + (size_t)i * inter * cin, X, BB, up[u].bb + (size_t)i * inter,
                 inter, P, cin, 0, sX, sAB, 0, 1.f, 0, 0);
            gemm(AA, BB, AT, nullptr, R, R, inter * D, sAB, sAB, (long long)R * R,
                 1, 1.f / (float)(inter * D), 0, 0);
            skg_softmax<<<dim3(SKG_S * R), dim3(32), 0, stream>>>(
                AT, (u < 10) ? (ASM + i * 625) : nullptr, R);
            gemm(X, AT, Z, nullptr, cin * D, R, R, sX, (long long)R * R, sX, 0, 1.f, 0, 0);
            gemm(up[u].wd + (size_t)i * cout * cin, Z, Y, up[u].bd + (size_t)i * cout,
                 cout, P, cin, 0, sX, sYc, 0, 1.f, (i > 0) ? 1 : 0, 0);
        }
        skg_bnstats<<<dim3(cout), dim3(256), 0, stream>>>(Y, MV, cout, P);
        const long long totG = (long long)SKG_S * cout * P;
        if (up[u].down_w) {
            gemm(up[u].down_w, X, Z, up[u].down_b, cout, P, cin, 0, sX, sYc, 0, 1.f, 0, 0);
            skg_bnstats<<<dim3(cout), dim3(256), 0, stream>>>(Z, MV2, cout, P);
            skg_bnapply<<<dim3(2048), dim3(256), 0, stream>>>(
                Y, MV, up[u].bn_g, up[u].bn_b, Z, MV2, up[u].dbn_g, up[u].dbn_b,
                G, cout, P, totG, 1);
        } else {
            skg_bnapply<<<dim3(2048), dim3(256), 0, stream>>>(
                Y, MV, up[u].bn_g, up[u].bn_b, X, nullptr, nullptr, nullptr,
                G, cout, P, totG, 1);
        }

        // ---- TCN (9x1, pad 4, stride) + residual + relu
        const int Dout = (D - 1) / c.stride + 1;
        const int Pout = Dout * R;
        {
            dim3 g((Pout + 63) / 64, (cout + 31) / 32, SKG_S);
            skg_tcn<<<g, dim3(128), 0, stream>>>(up[u].tcn_w, G, Y, up[u].tcn_b,
                cout, Pout, cout * 9, D, R, c.stride, 9, 4, sYc, (long long)cout * Pout);
        }
        skg_bnstats<<<dim3(cout), dim3(256), 0, stream>>>(Y, MV, cout, Pout);
        const long long totT = (long long)SKG_S * cout * Pout;
        if (c.res == 2) {
            dim3 g((Pout + 63) / 64, (cout + 31) / 32, SKG_S);
            skg_tcn<<<g, dim3(128), 0, stream>>>(up[u].res_w, X, Z, up[u].res_b,
                cout, Pout, cin, D, R, c.stride, 1, 0, sX, (long long)cout * Pout);
            skg_bnstats<<<dim3(cout), dim3(256), 0, stream>>>(Z, MV2, cout, Pout);
            skg_bnapply<<<dim3(2048), dim3(256), 0, stream>>>(
                Y, MV, up[u].tcn_g, up[u].tcn_beta, Z, MV2, up[u].res_g, up[u].res_beta,
                X, cout, Pout, totT, 1);
        } else if (c.res == 1) {
            skg_bnapply<<<dim3(2048), dim3(256), 0, stream>>>(
                Y, MV, up[u].tcn_g, up[u].tcn_beta, X, nullptr, nullptr, nullptr,
                X, cout, Pout, totT, 1);
        } else {
            skg_bnapply<<<dim3(2048), dim3(256), 0, stream>>>(
                Y, MV, up[u].tcn_g, up[u].tcn_beta, nullptr, nullptr, nullptr, nullptr,
                X, cout, Pout, totT, 1);
        }
        D = Dout;
    }

    // ---- global mean over (batch, D, R) then FC -> 12 logits
    skg_pool<<<dim3(256), dim3(256), 0, stream>>>(X, PO, 256, D * R);
    skg_fc<<<dim3(1), dim3(256), 0, stream>>>(PO, fc_w, fc_b, (float *)d_out);
}
